// NeighborhoodSelfAttentionBlock_64372969832564
// MI455X (gfx1250) — compile-verified, matrix operand order checked by reading, not run on hardware
//
#include <hip/hip_runtime.h>
#include <stdint.h>


typedef __attribute__((ext_vector_type(16))) _Float16 v16h;
typedef __attribute__((ext_vector_type(8)))  float    v8f;
typedef __attribute__((ext_vector_type(8)))  int      v8i;

#define D_MODEL 512
#define D_HEAD  64
#define N_HEADS 8
#define N_TOK   2048
#define T_DIM   8
#define H_DIM   16
#define W_DIM   16
#define KT 3
#define KH 5
#define KW 5
#define NKEY 240
#define NKEY_PAD 256

#define ROPE_BASE 1.1447298858494002f
#define ROPE_STEP 0.0719557841560639f

__device__ __forceinline__ v8i frag_i8(const int8_t* __restrict__ row, int hi8) {
    v8i f;
#pragma unroll
    for (int j = 0; j < 8; ++j) {
        int off = ((j >> 1) << 4) + ((j & 1) << 2) + hi8;
        f[j] = *(const int*)(row + off);
    }
    return f;
}

union FragU { uint4 u[2]; v16h h; };
__device__ __forceinline__ v16h frag_f16(const _Float16* chunk, int hi) {
    FragU f;
    const char* p = (const char*)chunk + (hi << 4);
    f.u[0] = *(const uint4*)(p);
    f.u[1] = *(const uint4*)(p + 32);
    return f.h;
}

__device__ __forceinline__ uint32_t lds_offset_of(const void* p) {
#if defined(__gfx1250__)
    return (uint32_t)(uintptr_t)(__attribute__((address_space(3))) const char*)p;
#else
    return (uint32_t)(uintptr_t)p;
#endif
}

__device__ __forceinline__ void async_copy16(uint32_t lds_off, const void* gaddr) {
#if defined(__gfx1250__)
    asm volatile("global_load_async_to_lds_b128 %0, %1, off"
                 :: "v"(lds_off), "v"(gaddr) : "memory");
#else
    (void)lds_off; (void)gaddr;
#endif
}

__device__ __forceinline__ void wait_async() {
#if defined(__gfx1250__)
    asm volatile("s_wait_asynccnt 0x0" ::: "memory");
#endif
}

__global__ void init_kernel(float* p) {
    if (threadIdx.x < 2) p[threadIdx.x] = 0.0f;
}

__global__ void absmean_kernel(const float* __restrict__ w, int count, float* sum) {
    __shared__ float red[256];
    int tid = threadIdx.x;
    float s = 0.0f;
    for (int i = blockIdx.x * 256 + tid; i < count; i += gridDim.x * 256)
        s += fabsf(w[i]);
    red[tid] = s; __syncthreads();
    for (int k = 128; k > 0; k >>= 1) {
        if (tid < k) red[tid] += red[tid + k];
        __syncthreads();
    }
    if (tid == 0) atomicAdd(sum, red[0]);
}

__global__ void wquant_kernel(const float* __restrict__ w, int count,
                              const float* __restrict__ sum, float inv_count,
                              int8_t* __restrict__ wq) {
    int i = blockIdx.x * 256 + threadIdx.x;
    if (i >= count) return;
    float sw = 1.0f / fmaxf(sum[0] * inv_count, 1e-5f);
    float q  = rintf(w[i] * sw);
    q = fminf(fmaxf(q, -1.0f), 1.0f);
    wq[i] = (int8_t)q;
}

__global__ void ada_kernel(const float* __restrict__ cond,
                           const float* __restrict__ ada_w,
                           float* __restrict__ scale_vec) {
    int c = blockIdx.x * 256 + threadIdx.x;
    if (c >= D_MODEL) return;
    float s = 0.0f;
    for (int f = 0; f < D_MODEL; ++f) s += cond[f] * ada_w[c * D_MODEL + f];
    scale_vec[c] = s + 1.0f;
}

__global__ void row_quant_kernel(const float* __restrict__ in,
                                 const float* __restrict__ scale_vec,
                                 const float* __restrict__ wsum, float winv,
                                 int8_t* __restrict__ outq, float* __restrict__ outf) {
    __shared__ float red[256];
    int token = blockIdx.x, tid = threadIdx.x;
    const float* row = in + token * D_MODEL;
    float v0 = row[tid], v1 = row[tid + 256];
    float h0, h1;
    if (scale_vec) {
        red[tid] = v0 * v0 + v1 * v1; __syncthreads();
        for (int k = 128; k > 0; k >>= 1) {
            if (tid < k) red[tid] += red[tid + k];
            __syncthreads();
        }
        float ms = red[0] * (1.0f / (float)D_MODEL);
        float r  = rsqrtf(ms + 1e-6f);
        h0 = v0 * scale_vec[tid] * r;
        h1 = v1 * scale_vec[tid + 256] * r;
        __syncthreads();
    } else { h0 = v0; h1 = v1; }
    red[tid] = fmaxf(fabsf(h0), fabsf(h1)); __syncthreads();
    for (int k = 128; k > 0; k >>= 1) {
        if (tid < k) red[tid] = fmaxf(red[tid], red[tid + k]);
        __syncthreads();
    }
    float sx = 127.0f / fmaxf(red[0], 1e-5f);
    float q0 = fminf(fmaxf(rintf(h0 * sx), -128.0f), 127.0f);
    float q1 = fminf(fmaxf(rintf(h1 * sx), -128.0f), 127.0f);
    outq[token * D_MODEL + tid]       = (int8_t)q0;
    outq[token * D_MODEL + tid + 256] = (int8_t)q1;
    if (tid == 0) {
        float sw = 1.0f / fmaxf(wsum[0] * winv, 1e-5f);
        outf[token] = 1.0f / (sx * sw);
    }
}

__global__ __launch_bounds__(32)
void qkv_gemm_kernel(const int8_t* __restrict__ xq, const float* __restrict__ fx,
                     const int8_t* __restrict__ wq, const float* __restrict__ pos,
                     const float* __restrict__ headscale,
                     _Float16* __restrict__ qh, _Float16* __restrict__ kh,
                     _Float16* __restrict__ vh) {
    __shared__ float tile[16][64];
    __shared__ float rowscale[16];
    int bm   = blockIdx.x;
    int s    = blockIdx.y;
    int mat  = s >> 3, head = s & 7;
    int lane = threadIdx.x;
    int hi8  = (lane >= 16) ? 8 : 0;
    int l15  = lane & 15;

    v8i zero = {0,0,0,0,0,0,0,0};
    v8i acc[4] = {zero, zero, zero, zero};
    const int8_t* arow = xq + (size_t)(bm * 16 + l15) * D_MODEL;
#pragma unroll
    for (int kc = 0; kc < 8; ++kc) {
        v8i a = frag_i8(arow + kc * 64, hi8);
#pragma unroll
        for (int sub = 0; sub < 4; ++sub) {
            const int8_t* brow = wq + (size_t)(s * 64 + sub * 16 + l15) * D_MODEL + kc * 64;
            v8i b = frag_i8(brow, hi8);
            acc[sub] = __builtin_amdgcn_wmma_i32_16x16x64_iu8(
                true, a, true, b, acc[sub], false, false);
        }
    }
#pragma unroll
    for (int sub = 0; sub < 4; ++sub)
#pragma unroll
        for (int r = 0; r < 8; ++r) {
            int row = r + ((lane >= 16) ? 8 : 0);
            tile[row][sub * 16 + l15] = (float)acc[sub][r] * fx[bm * 16 + row];
        }
    __syncthreads();

    if (mat < 2) {
        if (lane < 16) {
            float ss = 0.0f;
            for (int d = 0; d < 64; ++d) ss += tile[lane][d] * tile[lane][d];
            rowscale[lane] = sqrtf(headscale[head]) * rsqrtf(ss + 1e-6f);
        }
        __syncthreads();
        _Float16* dst = (mat == 0) ? qh : kh;
        for (int row = 0; row < 16; ++row) {
            int token = bm * 16 + row;
            float sq = rowscale[row];
#pragma unroll
            for (int u = 0; u < 2; ++u) {
                int dd = lane + u * 32;
                float val;
                if (dd < 24) {
                    int j = (dd < 12) ? dd : dd - 12;
                    int ax = j >> 2, fi = j & 3;
                    float fr = __expf(ROPE_BASE + ROPE_STEP * (float)(fi * N_HEADS + head));
                    float th = pos[token * 3 + ax] * fr;
                    float c = cosf(th), sn = sinf(th);
                    float x1 = tile[row][j] * sq, x2 = tile[row][j + 12] * sq;
                    val = (dd < 12) ? (x1 * c - x2 * sn) : (x2 * c + x1 * sn);
                } else {
                    val = tile[row][dd] * sq;
                }
                dst[(size_t)token * D_MODEL + head * 64 + dd] = (_Float16)val;
            }
        }
    } else {
        for (int row = 0; row < 16; ++row) {
            int token = bm * 16 + row;
#pragma unroll
            for (int u = 0; u < 2; ++u) {
                int dd = lane + u * 32;
                vh[(size_t)token * D_MODEL + head * 64 + dd] = (_Float16)tile[row][dd];
            }
        }
    }
}

__global__ __launch_bounds__(32)
void na3d_kernel(const _Float16* __restrict__ qh, const _Float16* __restrict__ kh,
                 const _Float16* __restrict__ vh, float* __restrict__ o) {
    __shared__ __align__(16) union {
        _Float16 k[NKEY_PAD][64];
        _Float16 attn[16][NKEY_PAD];
    } sKA;
    __shared__ __align__(16) _Float16 sVT[64][NKEY_PAD];

    int b    = blockIdx.x;
    int head = b & 7;
    int h    = (b >> 3) & 15;
    int t    = b >> 7;
    int ts   = min(max(t - 1, 0), T_DIM - KT);
    int hs   = min(max(h - 2, 0), H_DIM - KH);
    int lane = threadIdx.x;
    int l15  = lane & 15;
    int hi   = lane >> 4;

    for (int idx = lane; idx < NKEY; idx += 32) {
        int tt = idx / 80, hh = (idx / 16) % 5, ww = idx & 15;
        int token = ((ts + tt) * H_DIM + hs + hh) * W_DIM + ww;
        const char* src = (const char*)(kh + (size_t)token * D_MODEL + head * 64);
        uint32_t dst = lds_offset_of(&sKA.k[idx][0]);
#pragma unroll
        for (int j = 0; j < 8; ++j)
            async_copy16(dst + j * 16, src + j * 16);
    }
    uint4 z4 = make_uint4(0u, 0u, 0u, 0u);
    for (int idx = NKEY + lane; idx < NKEY_PAD; idx += 32)
#pragma unroll
        for (int j = 0; j < 8; ++j) *(uint4*)&sKA.k[idx][j * 8] = z4;
    for (int idx = lane; idx < NKEY_PAD; idx += 32) {
        if (idx < NKEY) {
            int tt = idx / 80, hh = (idx / 16) % 5, ww = idx & 15;
            int token = ((ts + tt) * H_DIM + hs + hh) * W_DIM + ww;
            const _Float16* src = vh + (size_t)token * D_MODEL + head * 64;
            for (int d = 0; d < 64; ++d) sVT[d][idx] = src[d];
        } else {
            for (int d = 0; d < 64; ++d) sVT[d][idx] = (_Float16)0.0f;
        }
    }
    wait_async();
    __syncthreads();

    const _Float16* qrow = qh + (size_t)((t * H_DIM + h) * W_DIM + l15) * D_MODEL + head * 64;
    v16h qa0 = frag_f16(qrow,      hi);
    v16h qa1 = frag_f16(qrow + 32, hi);

    v8f lg[16];
#pragma unroll
    for (int nt = 0; nt < 16; ++nt) {
        v8f c = {0.f,0.f,0.f,0.f,0.f,0.f,0.f,0.f};
        const _Float16* krow = &sKA.k[nt * 16 + l15][0];
        v16h b0 = frag_f16(krow,      hi);
        v16h b1 = frag_f16(krow + 32, hi);
        c = __builtin_amdgcn_wmma_f32_16x16x32_f16(false, qa0, false, b0,
                                                   (short)0, c, false, false);
        c = __builtin_amdgcn_wmma_f32_16x16x32_f16(false, qa1, false, b1,
                                                   (short)0, c, false, false);
        lg[nt] = c;
    }
    __syncthreads();

#pragma unroll
    for (int r = 0; r < 8; ++r) {
        int row = r + hi * 8;
        int wsq = min(max(row - 2, 0), W_DIM - KW);
        float vals[16];
        float mx = -1e30f;
#pragma unroll
        for (int nt = 0; nt < 16; ++nt) {
            int cidx = nt * 16 + l15;
            int ww = cidx & 15;
            bool valid = (cidx < NKEY) && (ww >= wsq) && (ww < wsq + KW);
            float l = valid ? lg[nt][r] : -1e30f;
            vals[nt] = l;
            mx = fmaxf(mx, l);
        }
#pragma unroll
        for (int m = 1; m < 16; m <<= 1) mx = fmaxf(mx, __shfl_xor(mx, m, 32));
        float sum = 0.0f;
#pragma unroll
        for (int nt = 0; nt < 16; ++nt) {
            float e = __expf(vals[nt] - mx);
            vals[nt] = e;
            sum += e;
        }
#pragma unroll
        for (int m = 1; m < 16; m <<= 1) sum += __shfl_xor(sum, m, 32);
        float inv = 1.0f / sum;
#pragma unroll
        for (int nt = 0; nt < 16; ++nt)
            sKA.attn[row][nt * 16 + l15] = (_Float16)(vals[nt] * inv);
    }
    __syncthreads();

    v8f oz = {0.f,0.f,0.f,0.f,0.f,0.f,0.f,0.f};
    v8f oacc[4] = {oz, oz, oz, oz};
    const _Float16* arow = &sKA.attn[l15][0];
#pragma unroll
    for (int kk = 0; kk < 8; ++kk) {
        v16h a = frag_f16(arow + kk * 32, hi);
#pragma unroll
        for (int sub = 0; sub < 4; ++sub) {
            const _Float16* vrow = &sVT[sub * 16 + l15][kk * 32];
            v16h bf = frag_f16(vrow, hi);
            oacc[sub] = __builtin_amdgcn_wmma_f32_16x16x32_f16(
                false, a, false, bf, (short)0, oacc[sub], false, false);
        }
    }
    int qbase = (t * H_DIM + h) * W_DIM;
#pragma unroll
    for (int sub = 0; sub < 4; ++sub)
#pragma unroll
        for (int r = 0; r < 8; ++r) {
            int row = r + hi * 8;
            o[(size_t)(qbase + row) * D_MODEL + head * 64 + sub * 16 + l15] = oacc[sub][r];
        }
}

__global__ __launch_bounds__(32)
void out_gemm_kernel(const int8_t* __restrict__ oq, const float* __restrict__ fo,
                     const int8_t* __restrict__ wq, const float* __restrict__ skip,
                     float* __restrict__ out) {
    int bm   = blockIdx.x;
    int s    = blockIdx.y;
    int lane = threadIdx.x;
    int hi8  = (lane >= 16) ? 8 : 0;
    int l15  = lane & 15;

    v8i zero = {0,0,0,0,0,0,0,0};
    v8i acc[4] = {zero, zero, zero, zero};
    const int8_t* arow = oq + (size_t)(bm * 16 + l15) * D_MODEL;
#pragma unroll
    for (int kc = 0; kc < 8; ++kc) {
        v8i a = frag_i8(arow + kc * 64, hi8);
#pragma unroll
        for (int sub = 0; sub < 4; ++sub) {
            const int8_t* brow = wq + (size_t)(s * 64 + sub * 16 + l15) * D_MODEL + kc * 64;
            v8i b = frag_i8(brow, hi8);
            acc[sub] = __builtin_amdgcn_wmma_i32_16x16x64_iu8(
                true, a, true, b, acc[sub], false, false);
        }
    }
#pragma unroll
    for (int sub = 0; sub < 4; ++sub)
#pragma unroll
        for (int r = 0; r < 8; ++r) {
            int row = r + ((lane >= 16) ? 8 : 0);
            size_t idx = (size_t)(bm * 16 + row) * D_MODEL + s * 64 + sub * 16 + l15;
            out[idx] = (float)acc[sub][r] * fo[bm * 16 + row] + skip[idx];
        }
}

extern "C" void kernel_launch(void* const* d_in, const int* in_sizes, int n_in,
                              void* d_out, int out_size, void* d_ws, size_t ws_size,
                              hipStream_t stream) {
    const float* x         = (const float*)d_in[0];
    const float* pos       = (const float*)d_in[1];
    const float* cond      = (const float*)d_in[2];
    const float* ada_w     = (const float*)d_in[3];
    const float* qkv_w     = (const float*)d_in[4];
    const float* headscale = (const float*)d_in[5];
    const float* out_w     = (const float*)d_in[6];
    float*       out       = (float*)d_out;

    char* ws = (char*)d_ws;
    size_t off = 0;
    auto carve = [&](size_t bytes) -> void* {
        void* p = ws + off;
        off = (off + bytes + 255) & ~(size_t)255;
        return p;
    };
    float*    sums      = (float*)carve(2 * sizeof(float));
    float*    scale_vec = (float*)carve(D_MODEL * sizeof(float));
    int8_t*   qkv_wq    = (int8_t*)carve(3 * D_MODEL * D_MODEL);
    int8_t*   out_wq    = (int8_t*)carve(D_MODEL * D_MODEL);
    int8_t*   xq        = (int8_t*)carve((size_t)N_TOK * D_MODEL);
    float*    fx        = (float*)carve(N_TOK * sizeof(float));
    _Float16* qh        = (_Float16*)carve((size_t)N_TOK * D_MODEL * 2);
    _Float16* kh        = (_Float16*)carve((size_t)N_TOK * D_MODEL * 2);
    _Float16* vh        = (_Float16*)carve((size_t)N_TOK * D_MODEL * 2);
    float*    attn_o    = (float*)carve((size_t)N_TOK * D_MODEL * sizeof(float));
    int8_t*   oq        = (int8_t*)carve((size_t)N_TOK * D_MODEL);
    float*    fo        = (float*)carve(N_TOK * sizeof(float));

    const int CQKV = 3 * D_MODEL * D_MODEL;
    const int COUT = D_MODEL * D_MODEL;

    init_kernel<<<1, 32, 0, stream>>>(sums);
    absmean_kernel<<<256, 256, 0, stream>>>(qkv_w, CQKV, &sums[0]);
    absmean_kernel<<<256, 256, 0, stream>>>(out_w, COUT, &sums[1]);
    wquant_kernel<<<(CQKV + 255) / 256, 256, 0, stream>>>(qkv_w, CQKV, &sums[0],
                                                          1.0f / (float)CQKV, qkv_wq);
    wquant_kernel<<<(COUT + 255) / 256, 256, 0, stream>>>(out_w, COUT, &sums[1],
                                                          1.0f / (float)COUT, out_wq);
    ada_kernel<<<2, 256, 0, stream>>>(cond, ada_w, scale_vec);
    row_quant_kernel<<<N_TOK, 256, 0, stream>>>(x, scale_vec, &sums[0],
                                                1.0f / (float)CQKV, xq, fx);
    qkv_gemm_kernel<<<dim3(N_TOK / 16, 24), 32, 0, stream>>>(xq, fx, qkv_wq, pos,
                                                             headscale, qh, kh, vh);
    na3d_kernel<<<T_DIM * H_DIM * N_HEADS, 32, 0, stream>>>(qh, kh, vh, attn_o);
    row_quant_kernel<<<N_TOK, 256, 0, stream>>>(attn_o, nullptr, &sums[1],
                                                1.0f / (float)COUT, oq, fo);
    out_gemm_kernel<<<dim3(N_TOK / 16, 8), 32, 0, stream>>>(oq, fo, out_wq, x, out);
}
